// BNAF_33062658245364
// MI455X (gfx1250) — matrix operand order repaired and checked by reading, not compiled, not run on hardware
//
#include <hip/hip_runtime.h>
#include <hip/hip_bf16.h>
#include <math.h>

typedef _Float16 f16;
typedef __attribute__((ext_vector_type(16))) _Float16 v16h;
typedef __attribute__((ext_vector_type(8)))  _Float16 v8h;
typedef __attribute__((ext_vector_type(8)))  float    v8f;
typedef __attribute__((ext_vector_type(4)))  unsigned int u32x4;
typedef __attribute__((ext_vector_type(8)))  unsigned int u32x8;

#define BN 2048
#define HN 512
#define DN 8

// Branch-free fused tanh + tanh-log-det sharing one exp:
//   e = exp(-2|x|);  tanh(x) = copysign((1-e)/(1+e), x)
//   softplus(-2x) = max(-2x,0) + log(1+e);  ld = -2*(x - ln2 + softplus(-2x))
__device__ __forceinline__ void tanh_pair(float x, float& th, float& ld) {
    float e  = __expf(-2.f * fabsf(x));
    float t  = (1.f - e) * __builtin_amdgcn_rcpf(1.f + e);
    th       = __builtin_copysignf(t, x);
    float sp = fmaxf(-2.f * x, 0.f) + __logf(1.f + e);
    ld       = -2.f * (x - 0.6931471805599453f + sp);
}

// ---------------- per-layer weight-norm precompute (batch independent) ----------------
__global__ void prep_kernel(const float* __restrict__ W, const float* __restrict__ logg,
                            int in_f, int ob, int ib,
                            float* __restrict__ wf32, f16* __restrict__ wf16,
                            float* __restrict__ ldb) {
    const int row  = blockIdx.x;
    const int lane = threadIdx.x;              // 32 threads = 1 wave32
    const int blk  = row / ob;
    const int o    = row - blk * ob;
    const int dlo  = blk * ib, dhi = dlo + ib;
    const float lg = logg[row];

    float ss = 0.f;
    for (int c = lane; c < in_f; c += 32) {
        float Wv = W[row * in_f + c];
        float v  = (c < dlo) ? Wv : ((c < dhi) ? __expf(Wv) : 0.f);
        ss += v * v;
    }
    for (int off = 16; off > 0; off >>= 1) ss += __shfl_xor(ss, off, 32);

    float vnorm = sqrtf(ss);
    float scale = __expf(lg) / vnorm;
    float lvn   = __logf(vnorm);

    for (int c = lane; c < in_f; c += 32) {
        float Wv = W[row * in_f + c];
        float v  = (c < dlo) ? Wv : ((c < dhi) ? __expf(Wv) : 0.f);
        float wv = v * scale;
        if (wf32) wf32[row * in_f + c] = wv;
        if (wf16) wf16[row * in_f + c] = (f16)wv;
        if (c >= dlo && c < dhi)
            ldb[(blk * ob + o) * ib + (c - dlo)] = lg + Wv - lvn;
    }
}

// ---------------- layer 1: [B,8] x [512,8]^T, fused tanh/logdet, init sld ----------------
__global__ void layer1_kernel(const float* __restrict__ x, const float* __restrict__ w1,
                              const float* __restrict__ bias1, const float* __restrict__ ldb1,
                              f16* __restrict__ h1, float* __restrict__ sld) {
    int idx = blockIdx.x * blockDim.x + threadIdx.x;   // B*512
    int b = idx >> 9, r = idx & 511;
    float acc = bias1[r];
#pragma unroll
    for (int k = 0; k < DN; ++k) acc += x[b * DN + k] * w1[r * DN + k];
    float th, ld;
    tanh_pair(acc, th, ld);
    h1[idx]  = (f16)th;
    sld[idx] = ldb1[r] + ld;                 // lse over singleton k == ldb1
}

// ---------------- layers 2/3: 2048x512x512 GEMM via v_wmma_f32_16x16x32_f16 ----------------
// All 8 waves of a block share one 64-col weight strip, TDM-staged into LDS once.
// Waves differ in M-tile; each computes a 16(M) x 64(N) strip (4 accumulators).
// Block-causal weights are zero for k >= (nblock+1)*64 -> truncate K (exact).
__global__ void __launch_bounds__(256)
gemm_tanh_kernel(const f16* __restrict__ Xh, const f16* __restrict__ Wh,
                 const float* __restrict__ bias,
                 f16* __restrict__ Yh, float* __restrict__ Ldt) {
    __shared__ f16 s_b[64 * HN];   // 64 rows x up to 512 cols = 64KB, packed [64][kmax]

    const int nstrip = blockIdx.x & 7;                   // 0..7 (64-col weight strip)
    const int mgrp   = blockIdx.x >> 3;                  // 0..15
    const int wave   = threadIdx.x >> 5;                 // 0..7
    const int lane   = threadIdx.x & 31;
    const int m0     = (mgrp * 8 + wave) * 16;
    const int n0     = nstrip * 64;
    const int half   = lane >> 4;                        // K-half selector (contiguous k = 16*half + s, used for A and B)
    const int l16    = lane & 15;
    const int kmax   = (nstrip + 1) * 64;                // causal K truncation (uniform)

    // ---- Tensor Data Mover: stage 64 x kmax f16 strip (rows n0..n0+63) into LDS ----
    if (wave == 0) {
        unsigned long long gaddr = (unsigned long long)(uintptr_t)(Wh + (size_t)n0 * HN);
        unsigned int laddr = (unsigned int)(uintptr_t)(&s_b[0]);   // flat low 32 = LDS offset
        u32x4 g0;
        g0.x = 1u;                                                  // count=1 (valid D#)
        g0.y = laddr;                                               // lds_addr
        g0.z = (unsigned int)gaddr;                                 // global_addr[31:0]
        g0.w = ((unsigned int)(gaddr >> 32) & 0x1FFFFFFu) | (2u << 30); // addr[56:32] | type=2
        u32x8 g1;
        g1.s0 = 1u << 16;                    // wg_mask=0, data_size=1 (2 bytes), no flags
        g1.s1 = 512u << 16;                  // tensor_dim0[15:0] << 16  (dim0 = 512)
        g1.s2 = 512u << 16;                  // dim0 hi16 = 0 | tensor_dim1[15:0] << 16
        g1.s3 = ((unsigned int)kmax) << 16;  // dim1 hi16 = 0 | tile_dim0 = kmax
        g1.s4 = 64u;                         // tile_dim1 = 64, tile_dim2 = 0
        g1.s5 = 512u;                        // tensor_dim0_stride[31:0] = 512 elements
        g1.s6 = 0u;                          // stride0 hi | stride1 lo
        g1.s7 = 0u;                          // stride1 hi
        asm volatile("tensor_load_to_lds %0, %1" :: "s"(g0), "s"(g1) : "memory");
        __builtin_amdgcn_s_wait_tensorcnt(0);
    }
    __syncthreads();

    v8f c0 = {}, c1 = {}, c2 = {}, c3 = {};

    // A (16x32 f16): lane l16 = row; 16 contiguous halfs @ k+half*16
    const f16* arow = Xh + (m0 + l16) * HN + half * 16;
    // B (32x16 f16): lane l16 = col n; 16 contiguous halfs @ k+half*16 (from LDS strip)
    const f16* sb0 = s_b + ( 0 + l16) * kmax + half * 16;
    const f16* sb1 = s_b + (16 + l16) * kmax + half * 16;
    const f16* sb2 = s_b + (32 + l16) * kmax + half * 16;
    const f16* sb3 = s_b + (48 + l16) * kmax + half * 16;

    for (int kk = 0; kk < kmax; kk += 32) {
        v8h lo = *(const v8h*)(arow + kk);
        v8h hi = *(const v8h*)(arow + kk + 8);
        v16h a = __builtin_shufflevector(lo, hi, 0,1,2,3,4,5,6,7,8,9,10,11,12,13,14,15);
        v16h b0 = *(const v16h*)(sb0 + kk);
        v16h b1 = *(const v16h*)(sb1 + kk);
        v16h b2 = *(const v16h*)(sb2 + kk);
        v16h b3 = *(const v16h*)(sb3 + kk);
        c0 = __builtin_amdgcn_wmma_f32_16x16x32_f16(false, a, false, b0, (short)0, c0, false, false);
        c1 = __builtin_amdgcn_wmma_f32_16x16x32_f16(false, a, false, b1, (short)0, c1, false, false);
        c2 = __builtin_amdgcn_wmma_f32_16x16x32_f16(false, a, false, b2, (short)0, c2, false, false);
        c3 = __builtin_amdgcn_wmma_f32_16x16x32_f16(false, a, false, b3, (short)0, c3, false, false);
    }

    // epilogue: C/D layout -> lane holds N=l16, rows m0+half*8+r
#pragma unroll
    for (int t = 0; t < 4; ++t) {
        v8f cv = (t == 0) ? c0 : (t == 1) ? c1 : (t == 2) ? c2 : c3;
        int n = n0 + t * 16 + l16;
        float bv = bias[n];
#pragma unroll
        for (int r = 0; r < 8; ++r) {
            int m = m0 + half * 8 + r;
            float xv = cv[r] + bv;
            float th, ld;
            tanh_pair(xv, th, ld);
            Yh[m * HN + n]  = (f16)th;
            Ldt[m * HN + n] = ld;
        }
    }
}

// ---------------- sld logsumexp update: new[b,i,o] = lse_k(old[b,i,k]+ldb[i,o,k]) + ldt ----------------
__global__ void __launch_bounds__(256)
sld_kernel(const float* __restrict__ sld_in, const float* __restrict__ ldb,
           const float* __restrict__ ldt, float* __restrict__ sld_out) {
    __shared__ float s_ldb[64 * 64];   // stored transposed [k][o] -> conflict-free reads
    __shared__ float s_sld[4 * 64];
    const int i   = blockIdx.y;        // block index 0..7
    const int b0  = blockIdx.x * 4;
    const int tid = threadIdx.x;

    for (int t = tid; t < 4096; t += 256) {     // contiguous global reads, transpose to LDS
        int o = t >> 6, k = t & 63;
        s_ldb[k * 64 + o] = ldb[i * 4096 + t];
    }
    {
        int bl = tid >> 6, k = tid & 63;
        s_sld[tid] = sld_in[(b0 + bl) * HN + i * 64 + k];
    }
    __syncthreads();

    const int o = tid & 63, bl = tid >> 6;
    const int b = b0 + bl;
    float m = -INFINITY;
#pragma unroll 4
    for (int k = 0; k < 64; ++k) m = fmaxf(m, s_sld[bl * 64 + k] + s_ldb[k * 64 + o]);
    float s = 0.f;
#pragma unroll 4
    for (int k = 0; k < 64; ++k) s += __expf(s_sld[bl * 64 + k] + s_ldb[k * 64 + o] - m);
    sld_out[b * HN + i * 64 + o] = m + __logf(s) + ldt[b * HN + i * 64 + o];
}

// ---------------- layer 4: [B,512] x [8,512]^T + final tanh + final lse ----------------
__global__ void layer4_kernel(const f16* __restrict__ h3, const float* __restrict__ w4,
                              const float* __restrict__ bias4, const float* __restrict__ ldb4,
                              const float* __restrict__ sld_in, float* __restrict__ out) {
    int idx = blockIdx.x * blockDim.x + threadIdx.x;   // B*8
    int b = idx >> 3, r = idx & 7;
    int kmax = (r + 1) * 64;                           // causal truncation
    float acc = bias4[r];
    for (int k = 0; k < kmax; ++k) acc += (float)h3[b * HN + k] * w4[r * HN + k];
    float th, ld;
    tanh_pair(acc, th, ld);
    float m = -INFINITY;
    for (int k = 0; k < 64; ++k) m = fmaxf(m, sld_in[b * HN + r * 64 + k] + ldb4[r * 64 + k]);
    float s = 0.f;
    for (int k = 0; k < 64; ++k) s += __expf(sld_in[b * HN + r * 64 + k] + ldb4[r * 64 + k] - m);
    out[b * DN + r]           = th;                    // h
    out[BN * DN + b * DN + r] = m + __logf(s) + ld;    // sld
}

extern "C" void kernel_launch(void* const* d_in, const int* in_sizes, int n_in,
                              void* d_out, int out_size, void* d_ws, size_t ws_size,
                              hipStream_t stream) {
    (void)in_sizes; (void)n_in; (void)out_size; (void)ws_size;
    const float* x     = (const float*)d_in[0];
    const float* W1    = (const float*)d_in[1];
    const float* logg1 = (const float*)d_in[2];
    const float* bias1 = (const float*)d_in[3];
    const float* W2    = (const float*)d_in[4];
    const float* logg2 = (const float*)d_in[5];
    const float* bias2 = (const float*)d_in[6];
    const float* W3    = (const float*)d_in[7];
    const float* logg3 = (const float*)d_in[8];
    const float* bias3 = (const float*)d_in[9];
    const float* W4    = (const float*)d_in[10];
    const float* logg4 = (const float*)d_in[11];
    const float* bias4 = (const float*)d_in[12];

    char* p = (char*)d_ws;
    auto carve = [&](size_t bytes) -> void* {
        void* r = (void*)p;
        p += (bytes + 255) & ~(size_t)255;
        return r;
    };
    float* w1f  = (float*)carve(512 * 8 * 4);
    float* w4f  = (float*)carve(8 * 512 * 4);
    f16*   w2h  = (f16*)  carve(512 * 512 * 2);
    f16*   w3h  = (f16*)  carve(512 * 512 * 2);
    float* ldb1 = (float*)carve(512 * 4);
    float* ldb2 = (float*)carve(8 * 64 * 64 * 4);
    float* ldb3 = (float*)carve(8 * 64 * 64 * 4);
    float* ldb4 = (float*)carve(8 * 64 * 4);
    f16*   h1   = (f16*)  carve((size_t)BN * HN * 2);
    f16*   h2   = (f16*)  carve((size_t)BN * HN * 2);
    float* ldt  = (float*)carve((size_t)BN * HN * 4);
    float* sldA = (float*)carve((size_t)BN * HN * 4);
    float* sldB = (float*)carve((size_t)BN * HN * 4);
    f16*   h3   = h1;   // h1 consumed by GEMM2 before GEMM3 writes h3

    // batch-independent weight-norm + block logdets
    prep_kernel<<<512, 32, 0, stream>>>(W1, logg1,   8, 64,  1, w1f, nullptr, ldb1);
    prep_kernel<<<512, 32, 0, stream>>>(W2, logg2, 512, 64, 64, nullptr, w2h, ldb2);
    prep_kernel<<<512, 32, 0, stream>>>(W3, logg3, 512, 64, 64, nullptr, w3h, ldb3);
    prep_kernel<<<  8, 32, 0, stream>>>(W4, logg4, 512,  1, 64, w4f, nullptr, ldb4);

    // layer 1 (K=8, VALU) -> h1 (f16), sldA
    layer1_kernel<<<(BN * HN) / 256, 256, 0, stream>>>(x, w1f, bias1, ldb1, h1, sldA);

    // layer 2: TDM-staged WMMA GEMM + fused tanh/logdet; then sld logsumexp update
    gemm_tanh_kernel<<<128, 256, 0, stream>>>(h1, w2h, bias2, h2, ldt);
    sld_kernel<<<dim3(BN / 4, 8), 256, 0, stream>>>(sldA, ldb2, ldt, sldB);

    // layer 3
    gemm_tanh_kernel<<<128, 256, 0, stream>>>(h2, w3h, bias3, h3, ldt);
    sld_kernel<<<dim3(BN / 4, 8), 256, 0, stream>>>(sldB, ldb3, ldt, sldA);

    // layer 4 + outputs
    layer4_kernel<<<(BN * DN) / 256, 256, 0, stream>>>(h3, w4f, bias4, ldb4, sldA, (float*)d_out);
}
